// ChainAwareAttention_13675175870931
// MI455X (gfx1250) — compile-verified, matrix operand order checked by reading, not run on hardware
//
#include <hip/hip_runtime.h>
#include <hip/hip_bf16.h>

typedef _Float16 half_t;
typedef __attribute__((ext_vector_type(16))) _Float16 v16h;
typedef __attribute__((ext_vector_type(8)))  _Float16 v8h;
typedef __attribute__((ext_vector_type(4)))  _Float16 v4h;
typedef __attribute__((ext_vector_type(8)))  float    v8f;
typedef __attribute__((ext_vector_type(4)))  unsigned int u32x4;
typedef __attribute__((ext_vector_type(8)))  int          i32x8;
typedef __attribute__((ext_vector_type(4)))  int          i32x4;

#define D_MODEL  1024
#define N_HEADS  16
#define HEAD_DIM 64
#define SEQ      512
#define BATCH    4
#define M_TOTAL  (BATCH*SEQ)        // 2048
#define ATT_SCALE 0.125f            // 64^-0.5
#define ROPE_K   0.2878231366f      // ln(10000)/32

#if __has_builtin(__builtin_amdgcn_tensor_load_to_lds)
#define HAVE_TDM 1
#else
#define HAVE_TDM 0
#endif

// ---- fragment helpers -------------------------------------------------------
// Loads the per-lane 16-half fragment: 8 contiguous halves at p0 and 8 at p0+16.
// Works for both A-fragments (lane = row) and mirrored B-fragments (lane = col)
// when the source is row-major along the K dimension.
__device__ __forceinline__ v16h load_frag(const half_t* __restrict__ p0) {
  v8h lo = *(const v8h*)p0;
  v8h hi = *(const v8h*)(p0 + 16);
  v16h r;
#pragma unroll
  for (int i = 0; i < 8; ++i) { r[i] = lo[i]; r[8 + i] = hi[i]; }
  return r;
}

__device__ __forceinline__ v8f wmma_f16(v16h a, v16h b, v8f c) {
  return __builtin_amdgcn_wmma_f32_16x16x32_f16(
      /*neg_a=*/false, a, /*neg_b=*/false, b,
      /*c_mod=*/(short)0, c, /*reuse_a=*/false, /*reuse_b=*/false);
}

// ---- kernel 1: fused QKV GEMMs (6 matrices), f32 -> f16 cast in LDS,
//      RoPE fused for q_s/k_s, V stored transposed [b,h,d,s] ------------------
__global__ __launch_bounds__(256) void qkv_gemm(
    const float* __restrict__ x,
    const float* __restrict__ W0, const float* __restrict__ W1,
    const float* __restrict__ W2, const float* __restrict__ W3,
    const float* __restrict__ W4, const float* __restrict__ W5,
    half_t* __restrict__ o0, half_t* __restrict__ o1, half_t* __restrict__ o2,
    half_t* __restrict__ o3, half_t* __restrict__ o4, half_t* __restrict__ o5)
{
  __shared__ half_t Al[128][40];   // 128 rows x 32 k (padded stride 40 halves)
  __shared__ half_t Bl[64][40];    // 64 cols  x 32 k (transposed W tile)

  const int widx = blockIdx.z;
  const float* W = widx==0?W0: widx==1?W1: widx==2?W2: widx==3?W3: widx==4?W4: W5;
  half_t* out    = widx==0?o0: widx==1?o1: widx==2?o2: widx==3?o3: widx==4?o4: o5;
  const bool do_rope = (widx == 0) || (widx == 1);   // q_s, k_s
  const bool transp  = (widx == 2) || (widx == 5);   // v_s, v_c -> [b,h,d,s]

  const int tid = threadIdx.x;
  const int wv = tid >> 5, lane = tid & 31, lm = lane & 15, lh = lane >> 4;
  const int m0 = blockIdx.x * 128;
  const int h  = blockIdx.y;                 // BN = 64 = one head
  const int n0 = h * HEAD_DIM;

  v8f acc[4];
#pragma unroll
  for (int g = 0; g < 4; ++g) acc[g] = {};

  for (int k0 = 0; k0 < D_MODEL; k0 += 32) {
    // prefetch next K tiles into cache while we stage/compute this one
    if (k0 + 32 < D_MODEL) {
      __builtin_prefetch(x + (size_t)(m0 + (tid >> 3)) * D_MODEL + k0 + 32, 0, 3);
      __builtin_prefetch(W + (size_t)(k0 + 32 + (tid >> 4)) * D_MODEL + n0, 0, 3);
    }
    // stage A tile (x), f32 -> f16
#pragma unroll
    for (int i = 0; i < 4; ++i) {
      int id = tid + 256 * i;
      int row = id >> 3, kq = (id & 7) << 2;
      const float4 f = *(const float4*)(x + (size_t)(m0 + row) * D_MODEL + k0 + kq);
      v4h hv = { (half_t)f.x, (half_t)f.y, (half_t)f.z, (half_t)f.w };
      *(v4h*)&Al[row][kq] = hv;
    }
    // stage B tile (W), transposed to [n][k], f32 -> f16
#pragma unroll
    for (int i = 0; i < 2; ++i) {
      int id = tid + 256 * i;
      int kr = id >> 4, nq = (id & 15) << 2;
      const float4 f = *(const float4*)(W + (size_t)(k0 + kr) * D_MODEL + n0 + nq);
      Bl[nq + 0][kr] = (half_t)f.x;
      Bl[nq + 1][kr] = (half_t)f.y;
      Bl[nq + 2][kr] = (half_t)f.z;
      Bl[nq + 3][kr] = (half_t)f.w;
    }
    __syncthreads();
    v16h a = load_frag(&Al[wv * 16 + lm][lh * 8]);
#pragma unroll
    for (int g = 0; g < 4; ++g) {
      v16h b = load_frag(&Bl[g * 16 + lm][lh * 8]);
      acc[g] = wmma_f16(a, b, acc[g]);
    }
    __syncthreads();
  }

  // epilogue: optional RoPE (pairs d <-> d+32 are lane-local: acc g and g+2)
#pragma unroll
  for (int r = 0; r < 8; ++r) {
    int m = m0 + wv * 16 + r + 8 * lh;
    int b = m >> 9, sp = m & 511;
    if (do_rope) {
#pragma unroll
      for (int g = 0; g < 2; ++g) {
        int d = 16 * g + lm;                     // d in [0,32)
        float th = (float)sp * __expf(-(float)d * ROPE_K);
        float c = __cosf(th), sn = __sinf(th);
        float a0 = acc[g][r], b0 = acc[g + 2][r];
        acc[g][r]     = a0 * c - b0 * sn;
        acc[g + 2][r] = b0 * c + a0 * sn;
      }
    }
#pragma unroll
    for (int g = 0; g < 4; ++g) {
      int d = 16 * g + lm;
      half_t v = (half_t)acc[g][r];
      if (transp)
        out[(((size_t)b * N_HEADS + h) * HEAD_DIM + d) * SEQ + sp] = v;
      else
        out[(((size_t)b * N_HEADS + h) * SEQ + sp) * HEAD_DIM + d] = v;
    }
  }
}

// ---- kernel 2: flash-style chain-aware attention ----------------------------
// Computes S^T = K * Q^T so softmax stats are lane-local and P^T (C-layout)
// is directly the A-fragment for the PV WMMA.
__global__ __launch_bounds__(256) void attn(
    const half_t* __restrict__ qs, const half_t* __restrict__ ks,
    const half_t* __restrict__ vsT,
    const half_t* __restrict__ qc, const half_t* __restrict__ kc,
    const half_t* __restrict__ vcT,
    const int* __restrict__ chain_ids, const int* __restrict__ amask,
    half_t* __restrict__ attnh)
{
  __shared__ int   chainS[SEQ];
  __shared__ float padS[SEQ];
  const int bh = blockIdx.y, b = bh >> 4, h = bh & 15;
  const int tid = threadIdx.x, wv = tid >> 5, lane = tid & 31;
  const int lm = lane & 15, lh = lane >> 4;
  const int q0 = blockIdx.x * 128 + wv * 16;   // 16 queries per wave

  for (int i = tid; i < SEQ; i += 256) {
    chainS[i] = chain_ids[b * SEQ + i];
    padS[i]   = amask[b * SEQ + i] ? 0.0f : -1e30f;
  }
  __syncthreads();

  const size_t hoff = (size_t)bh * SEQ * HEAD_DIM;
  const half_t* Qs = qs  + hoff;
  const half_t* Ks = ks  + hoff;
  const half_t* Qc = qc  + hoff;
  const half_t* Kc = kc  + hoff;
  const half_t* Vs = vsT + hoff;   // [d][s] within head
  const half_t* Vc = vcT + hoff;

  // Q fragments (B-role, mirrored): lane = query column
  v16h qsf[2], qcf[2];
#pragma unroll
  for (int dg = 0; dg < 2; ++dg) {
    qsf[dg] = load_frag(Qs + (size_t)(q0 + lm) * HEAD_DIM + 32 * dg + lh * 8);
    qcf[dg] = load_frag(Qc + (size_t)(q0 + lm) * HEAD_DIM + 32 * dg + lh * 8);
  }
  const int cq = chainS[q0 + lm];

  v8f O[4];
#pragma unroll
  for (int g = 0; g < 4; ++g) O[g] = {};
  float m_run = -1e30f, ssum = 0.0f;

  for (int k0 = 0; k0 < SEQ; k0 += 32) {
    v8f S[2], C[2];
#pragma unroll
    for (int sub = 0; sub < 2; ++sub) {
      int kk = k0 + 16 * sub;
      v16h ka0 = load_frag(Ks + (size_t)(kk + lm) * HEAD_DIM + lh * 8);
      v16h ka1 = load_frag(Ks + (size_t)(kk + lm) * HEAD_DIM + 32 + lh * 8);
      v8f z = {};
      S[sub] = wmma_f16(ka0, qsf[0], z);
      S[sub] = wmma_f16(ka1, qsf[1], S[sub]);
      v16h ca0 = load_frag(Kc + (size_t)(kk + lm) * HEAD_DIM + lh * 8);
      v16h ca1 = load_frag(Kc + (size_t)(kk + lm) * HEAD_DIM + 32 + lh * 8);
      C[sub] = wmma_f16(ca0, qcf[0], z);
      C[sub] = wmma_f16(ca1, qcf[1], C[sub]);
    }
    // merge self/cross by chain id, online softmax (per query = per lane col)
    float vals[16]; bool intra[16];
    float mx = -1e30f;
#pragma unroll
    for (int sub = 0; sub < 2; ++sub) {
#pragma unroll
      for (int r = 0; r < 8; ++r) {
        int key = k0 + 16 * sub + r + 8 * lh;
        bool in = (chainS[key] == cq);
        float sc = (in ? S[sub][r] : C[sub][r]) * ATT_SCALE + padS[key];
        vals[sub * 8 + r] = sc; intra[sub * 8 + r] = in;
        mx = fmaxf(mx, sc);
      }
    }
    mx = fmaxf(mx, __shfl_xor(mx, 16, 32));
    float m_new = fmaxf(m_run, mx);
    float corr = __expf(m_run - m_new);
    float psum = 0.0f;
    v16h pSf, pCf;   // P^T C-layout == A-fragment layout, lane-local repack
#pragma unroll
    for (int e = 0; e < 16; ++e) {
      float p = __expf(vals[e] - m_new);
      psum += p;
      pSf[e] = intra[e] ? (half_t)p : (half_t)0.0f;
      pCf[e] = intra[e] ? (half_t)0.0f : (half_t)p;
    }
    ssum = ssum * corr + psum;
    m_run = m_new;
    // rescale O (rows = queries) with per-row correction gathered by shuffle
    float corrO[8];
#pragma unroll
    for (int r = 0; r < 8; ++r) corrO[r] = __shfl(corr, r + 8 * lh, 32);
#pragma unroll
    for (int g = 0; g < 4; ++g) {
#pragma unroll
      for (int r = 0; r < 8; ++r) O[g][r] *= corrO[r];
      v16h vsf = load_frag(Vs + (size_t)(16 * g + lm) * SEQ + k0 + lh * 8);
      O[g] = wmma_f16(pSf, vsf, O[g]);
      v16h vcf = load_frag(Vc + (size_t)(16 * g + lm) * SEQ + k0 + lh * 8);
      O[g] = wmma_f16(pCf, vcf, O[g]);
    }
  }

  float st = ssum + __shfl_xor(ssum, 16, 32);
  float inv = st > 0.0f ? 1.0f / st : 0.0f;
  float invO[8];
#pragma unroll
  for (int r = 0; r < 8; ++r) invO[r] = __shfl(inv, r + 8 * lh, 32);
#pragma unroll
  for (int r = 0; r < 8; ++r) {
    int q = q0 + r + 8 * lh;
#pragma unroll
    for (int g = 0; g < 4; ++g) {
      attnh[((size_t)(b * SEQ + q)) * D_MODEL + h * HEAD_DIM + 16 * g + lm] =
          (half_t)(O[g][r] * invO[r]);
    }
  }
}

// ---- kernel 3: output projection attnh(f16) @ Wo(f32->f16) -> f32 -----------
// The f16 A tile (128x32, row-major, stride 1024) is moved global->LDS by the
// Tensor Data Mover.  TDM padding (pad_interval=3: every 16 DWORDs, pad_amount=3:
// 4 DWORDs) reproduces the 40-half padded LDS row stride, so the compute-side
// fragment addressing is identical to the manual-staging path.
__global__ __launch_bounds__(256) void out_gemm(
    const half_t* __restrict__ A, const float* __restrict__ Wo,
    float* __restrict__ out)
{
  __shared__ half_t Al[128][40];
  __shared__ half_t Bl[64][40];
  const int tid = threadIdx.x, wv = tid >> 5, lane = tid & 31;
  const int lm = lane & 15, lh = lane >> 4;
  const int m0 = blockIdx.x * 128, n0 = blockIdx.y * 64;

#if HAVE_TDM
  const unsigned lds_a = (unsigned)(size_t)(&Al[0][0]);   // LDS byte offset
#endif

  v8f acc[4];
#pragma unroll
  for (int g = 0; g < 4; ++g) acc[g] = {};

  for (int k0 = 0; k0 < D_MODEL; k0 += 32) {
    if (k0 + 32 < D_MODEL)
      __builtin_prefetch(Wo + (size_t)(k0 + 32 + (tid >> 4)) * D_MODEL + n0, 0, 3);

#if HAVE_TDM
    if (wv == 0) {
      // D# group 0: count=1 | lds_addr | global_addr[56:0] | type=2
      unsigned long long ga =
          (unsigned long long)(size_t)(A + (size_t)m0 * D_MODEL + k0);
      u32x4 g0;
      g0[0] = 1u;                                   // count=1, user mode
      g0[1] = lds_a;                                // lds_addr (bytes)
      g0[2] = (unsigned)(ga & 0xFFFFFFFFu);         // global_addr[31:0]
      g0[3] = (unsigned)((ga >> 32) & 0x1FFFFFFu)   // global_addr[56:32]
            | 0x80000000u;                          // type=2 ("image")
      // D# group 1: data_size=2B, pad every 16 DWORDs by 4 DWORDs,
      // tensor 1024 x 2048 (stride 1024), tile 32 x 128
      i32x8 g1;
      g1[0] = (int)((1u << 16)        // data_size = 1 -> 2 bytes
                  | (1u << 20)        // pad_enable
                  | (3u << 22)        // pad_interval = 3 -> every 16 DWORDs
                  | (3u << 25));      // pad_amount  = 3 -> 4 DWORDs
      g1[1] = (int)(1024u << 16);                       // tensor_dim0 lo16
      g1[2] = (int)((2048u & 0xFFFFu) << 16);           // dim0 hi | dim1 lo
      g1[3] = (int)(32u << 16);                         // dim1 hi | tile_dim0=32
      g1[4] = 128;                                      // tile_dim1=128, dim2=0
      g1[5] = 1024;                                     // dim0_stride lo32
      g1[6] = (int)((2097152u & 0xFFFFu) << 16);        // stride hi | dim1_stride lo
      g1[7] = (int)(2097152u >> 16);                    // dim1_stride[47:16]
      i32x4 g2 = {0, 0, 0, 0}, g3 = {0, 0, 0, 0};
#if defined(__clang_major__) && __clang_major__ >= 23
      i32x8 gz = {0, 0, 0, 0, 0, 0, 0, 0};
      __builtin_amdgcn_tensor_load_to_lds(g0, g1, g2, g3, gz, 0);
#else
      __builtin_amdgcn_tensor_load_to_lds(g0, g1, g2, g3, 0);
#endif
    }
#else
    // fallback: manual A staging
#pragma unroll
    for (int i = 0; i < 2; ++i) {
      int id = tid + 256 * i;
      int row = id >> 2, kq = (id & 3) << 3;
      *(v8h*)&Al[row][kq] = *(const v8h*)(A + (size_t)(m0 + row) * D_MODEL + k0 + kq);
    }
#endif

    // B staging (Wo f32 -> f16, transposed) overlaps the TDM transfer
#pragma unroll
    for (int i = 0; i < 2; ++i) {
      int id = tid + 256 * i;
      int kr = id >> 4, nq = (id & 15) << 2;
      const float4 f = *(const float4*)(Wo + (size_t)(k0 + kr) * D_MODEL + n0 + nq);
      Bl[nq + 0][kr] = (half_t)f.x;
      Bl[nq + 1][kr] = (half_t)f.y;
      Bl[nq + 2][kr] = (half_t)f.z;
      Bl[nq + 3][kr] = (half_t)f.w;
    }
#if HAVE_TDM
    __builtin_amdgcn_s_wait_tensorcnt(0);
#endif
    __syncthreads();
    v16h a = load_frag(&Al[wv * 16 + lm][lh * 8]);
#pragma unroll
    for (int g = 0; g < 4; ++g) {
      v16h b = load_frag(&Bl[g * 16 + lm][lh * 8]);
      acc[g] = wmma_f16(a, b, acc[g]);
    }
    __syncthreads();
  }

#pragma unroll
  for (int r = 0; r < 8; ++r) {
    int m = m0 + wv * 16 + r + 8 * lh;
#pragma unroll
    for (int g = 0; g < 4; ++g)
      out[(size_t)m * D_MODEL + n0 + 16 * g + lm] = acc[g][r];
  }
}

// ---- launcher ---------------------------------------------------------------
extern "C" void kernel_launch(void* const* d_in, const int* in_sizes, int n_in,
                              void* d_out, int out_size, void* d_ws, size_t ws_size,
                              hipStream_t stream) {
  const float* x     = (const float*)d_in[0];
  const int*   chain = (const int*)d_in[1];
  const int*   amask = (const int*)d_in[2];
  const float* Wqs   = (const float*)d_in[3];
  const float* Wks   = (const float*)d_in[4];
  const float* Wvs   = (const float*)d_in[5];
  const float* Wqc   = (const float*)d_in[6];
  const float* Wkc   = (const float*)d_in[7];
  const float* Wvc   = (const float*)d_in[8];
  const float* Wo    = (const float*)d_in[9];
  float* out = (float*)d_out;

  const size_t ELT = (size_t)M_TOTAL * D_MODEL;   // 2M f16 elements per buffer
  half_t* ws   = (half_t*)d_ws;
  half_t* qsb  = ws + 0 * ELT;    // [b,h,s,d] f16, RoPE applied
  half_t* ksb  = ws + 1 * ELT;    // [b,h,s,d] f16, RoPE applied
  half_t* vsb  = ws + 2 * ELT;    // [b,h,d,s] f16 (transposed)
  half_t* qcb  = ws + 3 * ELT;
  half_t* kcb  = ws + 4 * ELT;
  half_t* vcb  = ws + 5 * ELT;
  half_t* atb  = ws + 6 * ELT;    // attention output, [b,s,h*d] f16

  qkv_gemm<<<dim3(16, 16, 6), 256, 0, stream>>>(
      x, Wqs, Wks, Wvs, Wqc, Wkc, Wvc, qsb, ksb, vsb, qcb, kcb, vcb);
  attn<<<dim3(4, 64), 256, 0, stream>>>(
      qsb, ksb, vsb, qcb, kcb, vcb, chain, amask, atb);
  out_gemm<<<dim3(16, 16), 256, 0, stream>>>(atb, Wo, out);
}
